// MultiQueryAttention_68599217652398
// MI455X (gfx1250) — matrix operand order repaired and checked by reading, not compiled, not run on hardware
//
#include <hip/hip_runtime.h>

#define Bc 2
#define Sc 4096
#define Ec 2048
#define Hc 16
#define HDc 128
#define QKD (Hc * HDc) /* 2048 */

typedef unsigned short u16;
typedef __attribute__((ext_vector_type(16))) __bf16 v16bf;
typedef __attribute__((ext_vector_type(8))) float v8f;
typedef __attribute__((ext_vector_type(4))) unsigned int v4u;

struct Frag32 {
  v4u lo;
  v4u hi;
};

static __device__ __forceinline__ u16 f32_to_bf16(float f) {
  unsigned int u = __builtin_bit_cast(unsigned int, f);
  u += 0x7FFFu + ((u >> 16) & 1u);  // round-to-nearest-even
  return (u16)(u >> 16);
}

// Load a 16-element bf16 WMMA fragment from two 16-byte chunks.
static __device__ __forceinline__ v16bf ld_frag2(const void* p0, const void* p1) {
  Frag32 f;
  f.lo = *(const v4u*)p0;
  f.hi = *(const v4u*)p1;
  return __builtin_bit_cast(v16bf, f);
}
// Load a 16-element bf16 fragment from one contiguous 32-byte chunk.
static __device__ __forceinline__ v16bf ld_frag1(const void* p) {
  Frag32 f;
  f.lo = ((const v4u*)p)[0];
  f.hi = ((const v4u*)p)[1];
  return __builtin_bit_cast(v16bf, f);
}

static __device__ __forceinline__ v8f wmma_bf16(v16bf a, v16bf b, v8f c) {
  return __builtin_amdgcn_wmma_f32_16x16x32_bf16(false, a, false, b, (short)0, c,
                                                 false, false);
}

#define LDS_FENCE() asm volatile("s_wait_dscnt 0" ::: "memory")

static __device__ __forceinline__ v8f v8f_zero() {
  v8f z = {0.f, 0.f, 0.f, 0.f, 0.f, 0.f, 0.f, 0.f};
  return z;
}

// ---------------- elementwise conversions ----------------

__global__ void k_cvt_bf16(const float* __restrict__ in, u16* __restrict__ out,
                           size_t n) {
  size_t i = (size_t)blockIdx.x * blockDim.x + threadIdx.x;
  if (i < n) out[i] = f32_to_bf16(in[i]);
}

// W[K][N] f32 (row-major) -> Wt[N][K] bf16 (row-major).
__global__ void k_cvt_wt(const float* __restrict__ W, u16* __restrict__ Wt,
                         int K, int N) {
  size_t i = (size_t)blockIdx.x * blockDim.x + threadIdx.x;
  size_t total = (size_t)K * N;
  if (i >= total) return;
  size_t n = i / (size_t)K;
  size_t kk = i - n * (size_t)K;
  Wt[i] = f32_to_bf16(W[kk * (size_t)N + n]);
}

// v[B*S][HD] bf16 -> vT[B][HD][S] bf16
__global__ void k_vt(const u16* __restrict__ v, u16* __restrict__ vT) {
  size_t i = (size_t)blockIdx.x * blockDim.x + threadIdx.x;
  size_t total = (size_t)Bc * HDc * Sc;
  if (i >= total) return;
  size_t b = i >> 19;  // HDc*Sc = 2^19
  size_t r = i & ((1u << 19) - 1);
  size_t hd = r >> 12;      // / Sc
  size_t s = r & (Sc - 1);  // % Sc
  vT[i] = v[((size_t)b * Sc + s) * HDc + hd];
}

// ---------------- generic bf16 WMMA GEMM ----------------
// C[M][N] = A[M][K](bf16) * Bt[N][K]^T(bf16) + bias[N]
// One wave: 32x64 output (2 m-tiles x 4 n-tiles), double-buffered k-chunks.
// Requires K % 64 == 0.

struct KStep {
  v16bf a[2];
  v16bf b[4];
};

static __device__ __forceinline__ void gemm_load(KStep& f, const u16* ap0,
                                                 const u16* ap1,
                                                 const u16* const bp[4],
                                                 int k) {
  f.a[0] = ld_frag1(ap0 + k);
  f.a[1] = ld_frag1(ap1 + k);
#pragma unroll
  for (int t = 0; t < 4; ++t) f.b[t] = ld_frag1(bp[t] + k);
}

static __device__ __forceinline__ void gemm_math(const KStep& f, v8f acc[2][4]) {
#pragma unroll
  for (int t = 0; t < 4; ++t) {
    acc[0][t] = wmma_bf16(f.a[0], f.b[t], acc[0][t]);
    acc[1][t] = wmma_bf16(f.a[1], f.b[t], acc[1][t]);
  }
}

__global__ __launch_bounds__(128) void k_gemm_bf16(
    const u16* __restrict__ A, const u16* __restrict__ Bt,
    const float* __restrict__ bias, void* __restrict__ Cout, int M, int N,
    int K, int out_f32) {
  const int lane = threadIdx.x & 31;
  const int wid = blockIdx.x * 4 + (threadIdx.x >> 5);
  const int nstrips = N >> 6;  // 64-wide strips
  const int mt = wid / nstrips;
  const int nt = wid - mt * nstrips;
  if (mt >= (M >> 5)) return;
  const int m0 = mt << 5, n0 = nt << 6;

  const int rowA = lane & 15;
  const int kb8 = (lane >> 4) * 8;    // C row offset (8 * lane half)
  const int kh16 = (lane >> 4) * 16;  // B-operand K half

  const u16* ap0 = A + (size_t)(m0 + rowA) * K + kh16;
  const u16* ap1 = A + (size_t)(m0 + 16 + rowA) * K + kh16;
  const u16* bp[4];
#pragma unroll
  for (int t = 0; t < 4; ++t)
    bp[t] = Bt + (size_t)(n0 + t * 16 + rowA) * K + kh16;

  v8f acc[2][4];
#pragma unroll
  for (int mi = 0; mi < 2; ++mi)
#pragma unroll
    for (int t = 0; t < 4; ++t) acc[mi][t] = v8f_zero();

  // software pipeline: preload chunk 0, then issue next chunk's loads
  // before consuming the current one (hides load latency behind 8 WMMAs).
  KStep f0, f1;
  gemm_load(f0, ap0, ap1, bp, 0);
  int k = 0;
  for (; k < K - 64; k += 64) {
    gemm_load(f1, ap0, ap1, bp, k + 32);
    gemm_math(f0, acc);
    gemm_load(f0, ap0, ap1, bp, k + 64);
    gemm_math(f1, acc);
  }
  gemm_load(f1, ap0, ap1, bp, k + 32);
  gemm_math(f0, acc);
  gemm_math(f1, acc);

#pragma unroll
  for (int mi = 0; mi < 2; ++mi) {
#pragma unroll
    for (int t = 0; t < 4; ++t) {
      const int col = n0 + t * 16 + rowA;
      const float bv = bias[col];
#pragma unroll
      for (int j = 0; j < 8; ++j) {
        const int row = m0 + mi * 16 + j + kb8;  // C: VGPR j -> row j+8*(l>>4)
        const float val = acc[mi][t][j] + bv;
        if (out_f32)
          ((float*)Cout)[(size_t)row * N + col] = val;
        else
          ((u16*)Cout)[(size_t)row * N + col] = f32_to_bf16(val);
      }
    }
  }
}

// ---------------- flash attention (causal, MQA) ----------------
// q: [B*S][QKD] bf16 (per-head slice), k: [B][S][HD] bf16,
// vT: [B][HD][S] bf16, o: [B*S][QKD] bf16.
// One wave = 16 query rows x full HD=128 for one (b,h).
__global__ __launch_bounds__(128) void k_attn(const u16* __restrict__ q,
                                              const u16* __restrict__ kmat,
                                              const u16* __restrict__ vT,
                                              u16* __restrict__ o,
                                              float scale) {
  const int lane = threadIdx.x & 31;
  const int warp = threadIdx.x >> 5;
  const int w = blockIdx.x * 4 + warp;
  const int qt = w & 255;       // S/16 query tiles
  const int h = (w >> 8) & 15;  // head
  const int b = w >> 12;        // batch

  __shared__ float s_scores[4][16 * 33];  // padded stride (bank conflicts)
  __shared__ __attribute__((aligned(32))) u16 s_probs[4][16 * 32];
  __shared__ float s_rowv[4][16];
  float* scores = s_scores[warp];
  u16* probs = s_probs[warp];
  float* rowv = s_rowv[warp];

  const u16* qb = q + (size_t)b * Sc * QKD + h * HDc;  // row stride QKD
  const u16* kb = kmat + (size_t)b * Sc * HDc;         // row stride HDc
  const u16* vb = vT + (size_t)b * HDc * Sc;           // [hd][s]

  const int q0 = qt << 4;
  const int rowA = lane & 15;
  const int kb8 = (lane >> 4) * 8;
  const int kh16 = (lane >> 4) * 16;

  // Q fragments: full HD=128 contraction = 4 k-steps of 32
  v16bf qf[4];
  {
    const u16* qrow = qb + (size_t)(q0 + rowA) * QKD;
#pragma unroll
    for (int t = 0; t < 4; ++t)
      qf[t] = ld_frag1(qrow + t * 32 + kh16);
  }

  v8f oacc[8];
#pragma unroll
  for (int d = 0; d < 8; ++d) oacc[d] = v8f_zero();

  float m_run = -1e30f;  // valid in lanes 0..15 (row = lane)
  float l_run = 0.f;

  const int qhi = q0 + 15;
  for (int kblk = 0; kblk <= qhi; kblk += 32) {
    // prefetch next key block's K rows (gfx1250 global_prefetch_b8)
    if (kblk + 32 <= qhi)
      __builtin_prefetch(kb + (size_t)(kblk + 32 + rowA) * HDc, 0, 0);

    // ---- scores: 16 queries x 32 keys (two 16x16 WMMA tiles) ----
    // Batch all 8 K-fragment loads first: one wait covers 8 WMMAs.
    v16bf kf[8];
    {
      const u16* krow0 = kb + (size_t)(kblk + rowA) * HDc + kh16;
      const u16* krow1 = kb + (size_t)(kblk + 16 + rowA) * HDc + kh16;
#pragma unroll
      for (int t = 0; t < 4; ++t) {
        kf[t] = ld_frag1(krow0 + t * 32);
        kf[4 + t] = ld_frag1(krow1 + t * 32);
      }
    }
    v8f sacc0 = v8f_zero(), sacc1 = v8f_zero();
#pragma unroll
    for (int t = 0; t < 4; ++t) {
      sacc0 = wmma_bf16(qf[t], kf[t], sacc0);
      sacc1 = wmma_bf16(qf[t], kf[4 + t], sacc1);
    }
#pragma unroll
    for (int j = 0; j < 8; ++j) {
      scores[(j + kb8) * 33 + rowA] = sacc0[j];
      scores[(j + kb8) * 33 + 16 + rowA] = sacc1[j];
    }

    // ---- issue V-fragment loads NOW: they depend only on kblk, so their
    // latency hides behind the whole LDS softmax round-trip below. ----
    v16bf vf[8];
#pragma unroll
    for (int d = 0; d < 8; ++d) {
      const int hd = d * 16 + rowA;  // B-operand column
      vf[d] = ld_frag1(vb + (size_t)hd * Sc + kblk + kh16);
    }

    LDS_FENCE();

    // ---- online softmax: lanes 0..15 each own one query row ----
    if (lane < 16) {
      const int r = lane;
      const int kmaxc = q0 + r - kblk;  // causal: local col c valid iff c<=kmaxc
      float m_new = m_run;
#pragma unroll
      for (int c = 0; c < 32; ++c) {
        float sv = scores[r * 33 + c] * scale;
        if (c <= kmaxc) m_new = fmaxf(m_new, sv);
      }
      const float alpha = __expf(m_run - m_new);
      float psum = 0.f;
#pragma unroll
      for (int c = 0; c < 32; ++c) {
        float sv = scores[r * 33 + c] * scale;
        float p = (c <= kmaxc) ? __expf(sv - m_new) : 0.f;
        psum += p;
        probs[r * 32 + c] = f32_to_bf16(p);
      }
      l_run = l_run * alpha + psum;
      m_run = m_new;
      rowv[r] = alpha;
    }
    LDS_FENCE();

    // ---- rescale accumulators by per-row alpha ----
#pragma unroll
    for (int j = 0; j < 8; ++j) {
      const float al = rowv[j + kb8];
#pragma unroll
      for (int d = 0; d < 8; ++d) oacc[d][j] *= al;
    }

    // ---- P(16x32, A layout from LDS probs) @ V(32xHD via vT rows) ----
    const u16* prow = probs + rowA * 32;
    v16bf pf = ld_frag1(prow + kh16);
#pragma unroll
    for (int d = 0; d < 8; ++d) oacc[d] = wmma_bf16(pf, vf[d], oacc[d]);
  }

  // ---- normalize by row sums and store bf16 ----
  if (lane < 16) rowv[lane] = 1.0f / l_run;
  LDS_FENCE();

  u16* ob = o + (size_t)(b * Sc + q0) * QKD + h * HDc;
#pragma unroll
  for (int j = 0; j < 8; ++j) {
    const float inv = rowv[j + kb8];
    u16* orow = ob + (size_t)(j + kb8) * QKD;
#pragma unroll
    for (int d = 0; d < 8; ++d)
      orow[d * 16 + rowA] = f32_to_bf16(oacc[d][j] * inv);
  }
}

// ---------------- host-side launch ----------------

extern "C" void kernel_launch(void* const* d_in, const int* in_sizes, int n_in,
                              void* d_out, int out_size, void* d_ws,
                              size_t ws_size, hipStream_t stream) {
  (void)in_sizes;
  (void)n_in;
  (void)out_size;
  (void)ws_size;
  const float* x = (const float*)d_in[0];
  const float* Wq = (const float*)d_in[1];
  const float* bq = (const float*)d_in[2];
  const float* Wk = (const float*)d_in[3];
  const float* bk = (const float*)d_in[4];
  const float* Wv = (const float*)d_in[5];
  const float* bv = (const float*)d_in[6];
  const float* Wo = (const float*)d_in[7];
  const float* bo = (const float*)d_in[8];
  float* out = (float*)d_out;

  char* ws = (char*)d_ws;
  const size_t M = (size_t)Bc * Sc;  // 8192
  size_t off = 0;
  u16* x_bf = (u16*)(ws + off);   off += M * Ec * 2;            // 32 MiB
  u16* Wqt = (u16*)(ws + off);    off += (size_t)QKD * Ec * 2;  // 8 MiB
  u16* Wkt = (u16*)(ws + off);    off += (size_t)HDc * Ec * 2;
  u16* Wvt = (u16*)(ws + off);    off += (size_t)HDc * Ec * 2;
  u16* Wot = (u16*)(ws + off);    off += (size_t)Ec * QKD * 2;  // 8 MiB
  u16* q_bf = (u16*)(ws + off);   off += M * QKD * 2;           // 32 MiB
  u16* k_bf = (u16*)(ws + off);   off += M * HDc * 2;
  u16* v_bf = (u16*)(ws + off);   off += M * HDc * 2;
  u16* vT_bf = (u16*)(ws + off);  off += M * HDc * 2;
  u16* at_bf = (u16*)(ws + off);  off += M * QKD * 2;           // 32 MiB

  const float scale = 0.08838834764831845f;  // 1/sqrt(128)

  // 1. conversions
  {
    size_t n = M * Ec;
    k_cvt_bf16<<<(unsigned)((n + 255) / 256), 256, 0, stream>>>(x, x_bf, n);
  }
  {
    size_t n = (size_t)Ec * QKD;
    k_cvt_wt<<<(unsigned)((n + 255) / 256), 256, 0, stream>>>(Wq, Wqt, Ec, QKD);
  }
  {
    size_t n = (size_t)Ec * HDc;
    k_cvt_wt<<<(unsigned)((n + 255) / 256), 256, 0, stream>>>(Wk, Wkt, Ec, HDc);
    k_cvt_wt<<<(unsigned)((n + 255) / 256), 256, 0, stream>>>(Wv, Wvt, Ec, HDc);
  }
  {
    size_t n = (size_t)QKD * Ec;
    k_cvt_wt<<<(unsigned)((n + 255) / 256), 256, 0, stream>>>(Wo, Wot, QKD, Ec);
  }

  // 2. QKV projections (bf16 out); 32x64 per wave, 4 waves/block
  {
    unsigned blocks = (unsigned)((M / 32) * (QKD / 64) / 4);  // 2048
    k_gemm_bf16<<<blocks, 128, 0, stream>>>(x_bf, Wqt, bq, q_bf, (int)M, QKD,
                                            Ec, 0);
  }
  {
    unsigned blocks = (unsigned)((M / 32) * (HDc / 64) / 4);  // 128
    k_gemm_bf16<<<blocks, 128, 0, stream>>>(x_bf, Wkt, bk, k_bf, (int)M, HDc,
                                            Ec, 0);
    k_gemm_bf16<<<blocks, 128, 0, stream>>>(x_bf, Wvt, bv, v_bf, (int)M, HDc,
                                            Ec, 0);
  }

  // 3. V transpose for the PV B-operand
  {
    size_t n = (size_t)Bc * HDc * Sc;
    k_vt<<<(unsigned)((n + 255) / 256), 256, 0, stream>>>(v_bf, vT_bf);
  }

  // 4. causal flash attention: B*H*(S/16) = 8192 waves, 4 waves/block
  k_attn<<<2048, 128, 0, stream>>>(q_bf, k_bf, vT_bf, at_bf, scale);

  // 5. output projection (f32 out into d_out)
  {
    unsigned blocks = (unsigned)((M / 32) * (Ec / 64) / 4);  // 2048
    k_gemm_bf16<<<blocks, 128, 0, stream>>>(at_bf, Wot, bo, out, (int)M, Ec,
                                            QKD, 1);
  }
}
